// SelfAttention_6751688589603
// MI455X (gfx1250) — matrix operand order repaired and checked by reading, not compiled, not run on hardware
//
#include <hip/hip_runtime.h>
#include <hip/hip_bf16.h>

// ---------------------------------------------------------------------------
// SelfAttention: out = softmax((xWq^T+bq)(xWk^T+bk)^T / sqrt(D)) @ x
// x: [SEQ, B, D] f32, out: [SEQ, B, D] f32.  SEQ=2048, B=8, D=1024.
// All matmuls via v_wmma_f32_16x16x32_bf16 (f32 accumulate).
// ---------------------------------------------------------------------------

#define SEQLEN 2048
#define NBATCH 8
#define DMODEL 1024

typedef __attribute__((ext_vector_type(16))) __bf16 v16bf;
typedef __attribute__((ext_vector_type(8)))  float  v8f;

union BfVec {
    v16bf v;
    uint4 q[2];
};

__device__ __forceinline__ unsigned short f2bf(float f) {
    unsigned int u = __float_as_uint(f);
    u += 0x7FFFu + ((u >> 16) & 1u);   // round-to-nearest-even
    return (unsigned short)(u >> 16);
}

// ---- workspace layout (ushort elements) -----------------------------------
// xb  [B][SEQ][D] bf16   @ 0              (16M elems, 32 MB)
// xbt [B][D][SEQ] bf16   @ 16M            (32 MB)
// wqb [D][D]      bf16   @ 32M            (2 MB)
// wkb [D][D]      bf16   @ 33M            (2 MB)
// qb  [B][SEQ][D] bf16   @ 34M            (32 MB)
// kb  [B][SEQ][D] bf16   @ 50M            (32 MB)
// total = 69,206,016 ushort = 132 MB  (requires ws_size >= 138412032 bytes)
static const size_t XB_OFF  = 0;
static const size_t XBT_OFF = 16777216UL;
static const size_t WQB_OFF = 33554432UL;
static const size_t WKB_OFF = 34603008UL;
static const size_t QB_OFF  = 35651584UL;
static const size_t KB_OFF  = 52428800UL;

// ---------------------------------------------------------------------------
// Kernel 1: convert weights f32 -> bf16
// ---------------------------------------------------------------------------
__global__ __launch_bounds__(256) void k_cvt_w(const float* __restrict__ Wq,
                                               const float* __restrict__ Wk,
                                               unsigned short* __restrict__ ws) {
    size_t idx = (size_t)blockIdx.x * 256 + threadIdx.x;   // 1M elements
    ws[WQB_OFF + idx] = f2bf(Wq[idx]);
    ws[WKB_OFF + idx] = f2bf(Wk[idx]);
}

// ---------------------------------------------------------------------------
// Kernel 2: x [SEQ,B,D] f32 -> xb [B][SEQ][D] bf16 (coalesced both sides)
// ---------------------------------------------------------------------------
__global__ __launch_bounds__(256) void k_cvt_x(const float* __restrict__ x,
                                               unsigned short* __restrict__ ws) {
    size_t idx = (size_t)blockIdx.x * 256 + threadIdx.x;   // 16M elements
    unsigned int d = (unsigned int)(idx & 1023u);
    unsigned int i = (unsigned int)((idx >> 10) & 2047u);
    unsigned int b = (unsigned int)(idx >> 21);
    ws[XB_OFF + idx] = f2bf(x[((size_t)(i * NBATCH + b) << 10) + d]);
}

// ---------------------------------------------------------------------------
// Kernel 3: build xbt [B][D][SEQ] bf16 via LDS-tiled 32x32 transpose of x
// ---------------------------------------------------------------------------
__global__ __launch_bounds__(256) void k_tr_x(const float* __restrict__ x,
                                              unsigned short* __restrict__ ws) {
    __shared__ unsigned short tile[32][33];
    int i0 = blockIdx.x * 32;   // seq tile
    int d0 = blockIdx.y * 32;   // d tile
    int b  = blockIdx.z;
    int tx = threadIdx.x & 31;
    int ty = threadIdx.x >> 5;  // 0..7
    for (int r = 0; r < 4; ++r) {
        int row = ty + r * 8;   // seq within tile
        float v = x[((size_t)(i0 + row) * NBATCH + b) * DMODEL + d0 + tx];
        tile[row][tx] = f2bf(v);
    }
    __syncthreads();
    unsigned short* xbt = ws + XBT_OFF;
    for (int r = 0; r < 4; ++r) {
        int drow = ty + r * 8;  // d within tile
        xbt[((size_t)b * DMODEL + d0 + drow) * SEQLEN + i0 + tx] = tile[tx][drow];
    }
}

// ---------------------------------------------------------------------------
// Kernel 4: projections Q = xWq^T + bq, K = xWk^T + bk  (bf16 WMMA, f32 acc)
// One wave per 16x16 output tile; Q and K share the A fragment.
// ---------------------------------------------------------------------------
__global__ __launch_bounds__(256) void k_proj(const float* __restrict__ bqv,
                                              const float* __restrict__ bkv,
                                              unsigned short* __restrict__ ws) {
    int wid  = (blockIdx.x * 256 + (int)threadIdx.x) >> 5;  // 65536 tiles
    int lane = threadIdx.x & 31;
    int b    = wid >> 13;           // 8192 tiles / batch
    int rem  = wid & 8191;
    int it   = rem >> 6;            // 0..127 seq tile
    int dt   = rem & 63;            // 0..63  d_out tile
    int m = lane & 15, h = lane >> 4;

    const unsigned short* xb  = ws + XB_OFF;
    const unsigned short* wqb = ws + WQB_OFF;
    const unsigned short* wkb = ws + WKB_OFF;
    const unsigned short* arow  = xb  + ((size_t)b * SEQLEN + it * 16 + m) * DMODEL;
    const unsigned short* bqrow = wqb + (size_t)(dt * 16 + m) * DMODEL;
    const unsigned short* bkrow = wkb + (size_t)(dt * 16 + m) * DMODEL;

    v8f cq = {}, ck = {};
    for (int kc = 0; kc < 32; ++kc) {
        int k0 = kc * 32;
        BfVec a, fq, fk;
        a.q[0] = *(const uint4*)(arow + k0 + h * 16);
        a.q[1] = *(const uint4*)(arow + k0 + h * 16 + 8);
        const unsigned short* pq = bqrow + k0 + h * 16;
        fq.q[0] = *(const uint4*)pq;
        fq.q[1] = *(const uint4*)(pq + 8);
        const unsigned short* pk = bkrow + k0 + h * 16;
        fk.q[0] = *(const uint4*)pk;
        fk.q[1] = *(const uint4*)(pk + 8);
        cq = __builtin_amdgcn_wmma_f32_16x16x32_bf16(false, a.v, false, fq.v,
                                                     (short)0, cq, false, false);
        ck = __builtin_amdgcn_wmma_f32_16x16x32_bf16(false, a.v, false, fk.v,
                                                     (short)0, ck, false, false);
    }
    int n = dt * 16 + m;
    float biasq = bqv[n], biask = bkv[n];
    unsigned short* qb = ws + QB_OFF;
    unsigned short* kb = ws + KB_OFF;
    for (int v = 0; v < 8; ++v) {
        int row = it * 16 + v + 8 * h;   // C layout: m = vgpr + 8*(lane>=16)
        qb[((size_t)b * SEQLEN + row) * DMODEL + n] = f2bf(cq[v] + biasq);
        kb[((size_t)b * SEQLEN + row) * DMODEL + n] = f2bf(ck[v] + biask);
    }
}

// ---------------------------------------------------------------------------
// Kernel 5: attention. One 8-wave workgroup per (batch, 16-row i-tile).
//   LDS: Q tile bf16 (padded rows), full score strip f32 [16][2048],
//        P bf16 (padded rows).
// ---------------------------------------------------------------------------
#define QROW 1032   // 1024 + 8 ushort pad: row stride 2064 B (16B aligned, 4-bank skew)
#define PROW 2056   // 2048 + 8 ushort pad: row stride 4112 B
#define LDS_Q_BYTES  (16 * QROW * 2)                  // 33024
#define LDS_S_OFF    LDS_Q_BYTES
#define LDS_S_BYTES  (16 * SEQLEN * 4)                // 131072
#define LDS_P_OFF    (LDS_S_OFF + LDS_S_BYTES)        // 164096
#define LDS_P_BYTES  (16 * PROW * 2)                  // 65792
#define LDS_TOTAL    (LDS_P_OFF + LDS_P_BYTES)        // 229888 bytes

__global__ __launch_bounds__(256, 1) void k_attn(const unsigned short* __restrict__ ws,
                                                 float* __restrict__ out) {
    extern __shared__ char smem[];
    unsigned short* qtile = (unsigned short*)smem;
    float*          sbuf  = (float*)(smem + LDS_S_OFF);
    unsigned short* pbuf  = (unsigned short*)(smem + LDS_P_OFF);

    int b  = blockIdx.x >> 7;
    int it = blockIdx.x & 127;
    int i0 = it * 16;
    int t = threadIdx.x, wave = t >> 5, lane = t & 31;
    int m = lane & 15, h = lane >> 4;

    const unsigned short* qb  = ws + QB_OFF;
    const unsigned short* kb  = ws + KB_OFF;
    const unsigned short* xbt = ws + XBT_OFF;

    // ---- stage Q tile (16 x 1024 bf16) into LDS, padded rows --------------
    const uint4* qsrc = (const uint4*)(qb + ((size_t)b * SEQLEN + i0) * DMODEL);
    for (int idx = t; idx < 2048; idx += 256) {       // 2048 x 16B
        int row = idx >> 7, col = idx & 127;
        ((uint4*)(qtile + row * QROW))[col] = qsrc[idx];
    }
    __syncthreads();

    // ---- phase 1: S = Q K^T * scale ---------------------------------------
    const float scale = 0.03125f;                     // 1/sqrt(1024)
    for (int jt = wave; jt < 128; jt += 8) {
        v8f c = {};
        const unsigned short* krow = kb + ((size_t)b * SEQLEN + jt * 16 + m) * DMODEL;
        const unsigned short* qrow = qtile + m * QROW;
        for (int kc = 0; kc < 32; ++kc) {
            int k0 = kc * 32;
            BfVec a, bf;
            a.q[0] = *(const uint4*)(qrow + k0 + h * 16);
            a.q[1] = *(const uint4*)(qrow + k0 + h * 16 + 8);
            const unsigned short* p = krow + k0 + h * 16;
            bf.q[0] = *(const uint4*)p;
            bf.q[1] = *(const uint4*)(p + 8);
            c = __builtin_amdgcn_wmma_f32_16x16x32_bf16(false, a.v, false, bf.v,
                                                        (short)0, c, false, false);
        }
        for (int v = 0; v < 8; ++v)
            sbuf[(v + 8 * h) * SEQLEN + jt * 16 + m] = c[v] * scale;
    }
    __syncthreads();

    // ---- phase 2: softmax over j, write P (normalized) as bf16 ------------
    {
        int row = t >> 4, s = t & 15;                 // 16 threads per row
        const float* srow = sbuf + row * SEQLEN;
        float mx = -3.402823466e38f;
        for (int col = s; col < SEQLEN; col += 16) mx = fmaxf(mx, srow[col]);
        for (int off = 8; off; off >>= 1) mx = fmaxf(mx, __shfl_xor(mx, off, 16));
        float l = 0.0f;
        for (int col = s; col < SEQLEN; col += 16) l += __expf(srow[col] - mx);
        for (int off = 8; off; off >>= 1) l += __shfl_xor(l, off, 16);
        float inv = 1.0f / l;
        unsigned short* prow = pbuf + row * PROW;
        for (int col = s; col < SEQLEN; col += 16)
            prow[col] = f2bf(__expf(srow[col] - mx) * inv);
    }
    __syncthreads();

    // ---- phase 3: O = P @ V  (V = x, read from transposed xbt) ------------
    for (int dt = wave; dt < 64; dt += 8) {
        v8f c = {};
        int n = dt * 16 + m;                          // output d column
        const unsigned short* vrow = xbt + ((size_t)b * DMODEL + n) * SEQLEN;
        const unsigned short* prow = pbuf + m * PROW;
        for (int jc = 0; jc < 64; ++jc) {
            int j0 = jc * 32;
            BfVec a, bf;
            a.q[0] = *(const uint4*)(prow + j0 + h * 16);
            a.q[1] = *(const uint4*)(prow + j0 + h * 16 + 8);
            const unsigned short* p = vrow + j0 + h * 16;
            bf.q[0] = *(const uint4*)p;
            bf.q[1] = *(const uint4*)(p + 8);
            c = __builtin_amdgcn_wmma_f32_16x16x32_bf16(false, a.v, false, bf.v,
                                                        (short)0, c, false, false);
        }
        for (int v = 0; v < 8; ++v) {
            int i = i0 + v + 8 * h;
            out[((size_t)i * NBATCH + b) * DMODEL + n] = c[v];
        }
    }
}

// ---------------------------------------------------------------------------
extern "C" void kernel_launch(void* const* d_in, const int* in_sizes, int n_in,
                              void* d_out, int out_size, void* d_ws, size_t ws_size,
                              hipStream_t stream) {
    const float* x  = (const float*)d_in[0];
    const float* Wq = (const float*)d_in[1];
    const float* bq = (const float*)d_in[2];
    const float* Wk = (const float*)d_in[3];
    const float* bk = (const float*)d_in[4];
    float* out = (float*)d_out;
    unsigned short* ws = (unsigned short*)d_ws;

    (void)in_sizes; (void)n_in; (void)out_size; (void)ws_size;

    // allow >64KB dynamic LDS for the attention kernel (idempotent host call)
    hipFuncSetAttribute((const void*)k_attn,
                        hipFuncAttributeMaxDynamicSharedMemorySize, LDS_TOTAL);

    // 1) weight conversion: 1M elements
    k_cvt_w<<<4096, 256, 0, stream>>>(Wq, Wk, ws);
    // 2) x -> xb bf16: 16M elements
    k_cvt_x<<<65536, 256, 0, stream>>>(x, ws);
    // 3) x -> xbt (transposed bf16)
    k_tr_x<<<dim3(SEQLEN / 32, DMODEL / 32, NBATCH), 256, 0, stream>>>(x, ws);
    // 4) Q/K projections: 65536 wave-tiles, 8 waves per block
    k_proj<<<8192, 256, 0, stream>>>(bq, bk, ws);
    // 5) attention: one workgroup per (batch, 16-row tile)
    k_attn<<<NBATCH * (SEQLEN / 16), 256, LDS_TOTAL, stream>>>(ws, out);
}
